// MinimalMambaBlock_76974403879713
// MI455X (gfx1250) — matrix operand order repaired and checked by reading, not compiled, not run on hardware
//
#include <hip/hip_runtime.h>
#include <stdint.h>

typedef __bf16 bf16;
typedef unsigned short u16;
typedef unsigned int   u32;
typedef __attribute__((ext_vector_type(8)))  bf16  v8bf;
typedef __attribute__((ext_vector_type(16))) bf16  v16bf;
typedef __attribute__((ext_vector_type(8)))  float v8f;
typedef __attribute__((ext_vector_type(4)))  float v4f;
typedef __attribute__((ext_vector_type(4)))  unsigned int u32x4;
typedef __attribute__((ext_vector_type(8)))  int i32x8;
typedef __attribute__((ext_vector_type(4)))  int i32x4;

#define DIM_    1024
#define INNER_  2048
#define NBATCH  4
#define SEQ     2048
#define NTOK    (NBATCH * SEQ)      // 8192 tokens
#define NCHUNK  8
#define CHUNK   (SEQ / NCHUNK)      // 256

// GEMM tiling: block 128x128, K-step 32, 8 waves (wave32), wave tile 32x64
#define BM   128
#define BN   128
#define BK   32
#define LDSK 40                      // padded LDS row stride (matches TDM pad: 16DW + 4DW)
#define TILE_ELEMS (BM * LDSK)       // 5120 bf16
#define TILE_BYTES (TILE_ELEMS * 2)  // 10240 B

// ---------------- bf16 helpers (bit-exact RNE, no target-feature deps) ----
static __device__ __forceinline__ u16 f2bf_bits(float f) {
    union { float f; u32 u; } c; c.f = f;
    u32 r = c.u + 0x7FFFu + ((c.u >> 16) & 1u);
    return (u16)(r >> 16);
}
static __device__ __forceinline__ bf16 f2bf(float f) {
    union { u16 h; bf16 b; } o; o.h = f2bf_bits(f);
    return o.b;
}
static __device__ __forceinline__ float bfbits2f(u32 hbits) {
    union { u32 u; float f; } o; o.u = hbits << 16;
    return o.f;
}

// ---------------- fp32 -> bf16 convert (weights) --------------------------
__global__ void k_f32_to_bf16(const float* __restrict__ in, bf16* __restrict__ out, int n4) {
    int i = blockIdx.x * blockDim.x + threadIdx.x;
    if (i < n4) {
        v4f v = ((const v4f*)in)[i];
        union { u16 h[4]; unsigned long long q; } p;
        p.h[0] = f2bf_bits(v.x); p.h[1] = f2bf_bits(v.y);
        p.h[2] = f2bf_bits(v.z); p.h[3] = f2bf_bits(v.w);
        ((unsigned long long*)out)[i] = p.q;
    }
}

// ---------------- RMSNorm: one block (256 thr) per token row --------------
__global__ __launch_bounds__(256) void k_rmsnorm(const float* __restrict__ x,
                                                 const float* __restrict__ nw,
                                                 bf16* __restrict__ xn) {
    const int row = blockIdx.x;
    const int tid = threadIdx.x;
    const float* xr = x + (size_t)row * DIM_;
    v4f v = ((const v4f*)xr)[tid];
    float ss = v.x * v.x + v.y * v.y + v.z * v.z + v.w * v.w;
    #pragma unroll
    for (int off = 16; off > 0; off >>= 1) ss += __shfl_xor(ss, off, 32);
    __shared__ float wsum[8];
    if ((tid & 31) == 0) wsum[tid >> 5] = ss;
    __syncthreads();
    float tot = wsum[0] + wsum[1] + wsum[2] + wsum[3] + wsum[4] + wsum[5] + wsum[6] + wsum[7];
    float scale = rsqrtf(tot * (1.0f / DIM_) + 1e-6f);
    v4f w = ((const v4f*)nw)[tid];
    bf16* o = xn + (size_t)row * DIM_ + tid * 4;
    union { u16 h[4]; unsigned long long q; } p;
    p.h[0] = f2bf_bits(v.x * scale * w.x);
    p.h[1] = f2bf_bits(v.y * scale * w.y);
    p.h[2] = f2bf_bits(v.z * scale * w.z);
    p.h[3] = f2bf_bits(v.w * scale * w.w);
    *(unsigned long long*)o = p.q;
}

// ---------------- WMMA fragment loads from LDS (ISA 7.12.2 layouts) -------
// A 16x32 bf16: lanes 0-15 row M=lane, K=0..7 & 16..23; lanes 16-31: K=8..15 & 24..31
static __device__ __forceinline__ v16bf fragA(const bf16* lds, int row0, int lane) {
    int r  = row0 + (lane & 15);
    int kb = (lane & 16) ? 8 : 0;
    const bf16* p = lds + r * LDSK + kb;
    v8bf lo = *(const v8bf*)p;
    v8bf hi = *(const v8bf*)(p + 16);
    return __builtin_shufflevector(lo, hi, 0,1,2,3,4,5,6,7,8,9,10,11,12,13,14,15);
}
// B 32x16 bf16 (stored [N,K] row-major): col N=lane&15; lanes 0-15 K=0..7 & 16..23, lanes 16-31 K=8..15 & 24..31
static __device__ __forceinline__ v16bf fragB(const bf16* lds, int row0, int lane) {
    int r  = row0 + (lane & 15);
    int kb = (lane & 16) ? 8 : 0;
    const bf16* p = lds + r * LDSK + kb;
    v8bf lo = *(const v8bf*)p;
    v8bf hi = *(const v8bf*)(p + 16);
    return __builtin_shufflevector(lo, hi, 0,1,2,3,4,5,6,7,8,9,10,11,12,13,14,15);
}

// ---------------- TDM: one descriptor loads a whole 128xBK tile -----------
// D# per CDNA5 ISA. 2-D tensor, data_size=2B,
// LDS padding: every 16 DWORDs (64B row) insert 4 DWORDs -> row stride 80B = LDSK*2.
static __device__ __forceinline__ void tdm_load_tile(const bf16* g, u32 ldsByteOff, int K) {
    unsigned long long ga = (unsigned long long)(uintptr_t)g;
    u32x4 g0;
    g0[0] = 1u;                               // count=1 valid descriptor
    g0[1] = ldsByteOff;                       // lds_addr (bytes)
    g0[2] = (u32)ga;                          // global_addr[31:0]
    g0[3] = (u32)(ga >> 32) | (2u << 30);     // global_addr[56:32] | type=2 (image)
    const u32 dim0 = (u32)K;                  // tensor_dim0 (elements)
    const u32 dim1 = (u32)BM;                 // tensor_dim1 (rows)
    i32x8 g1;
    g1[0] = (int)((1u << 16)                  // data_size = 2 bytes
                | (1u << 20)                  // pad_enable
                | (3u << 22)                  // pad_interval: 16 DWORDs
                | (3u << 25));                // pad_amount: 4 DWORDs
    g1[1] = (int)((dim0 & 0xFFFFu) << 16);               // tensor_dim0[15:0] @bits63:48
    g1[2] = (int)((dim0 >> 16) | ((dim1 & 0xFFFFu) << 16)); // dim0[31:16] | dim1[15:0]
    g1[3] = (int)((dim1 >> 16) | ((u32)BK << 16));       // dim1[31:16] | tile_dim0=32
    g1[4] = (int)BM;                                     // tile_dim1=128, tile_dim2=0
    g1[5] = (int)dim0;                                   // tensor_dim0_stride[31:0] = K
    g1[6] = 0;                                           // stride hi | dim1_stride lo
    g1[7] = 0;
    const i32x4 gz4 = {0, 0, 0, 0};                      // groups 2/3: 2-D tensor
    const i32x8 gz8 = {0, 0, 0, 0, 0, 0, 0, 0};          // unused trailing group
    __builtin_amdgcn_tensor_load_to_lds(g0, g1, gz4, gz4, gz8, 0);
}

// ---------------- templated WMMA GEMM:  C[m,n] = sum_k A[m,k]*B[n,k] ------
enum { EPI_GATED = 0, EPI_BF16 = 1, EPI_RESID = 2 };

template <int EPI>
__global__ __launch_bounds__(256) void k_gemm(
    const bf16* __restrict__ A, const bf16* __restrict__ B1, const bf16* __restrict__ B2,
    const float* __restrict__ bias1, const float* __restrict__ bias2,
    const float* __restrict__ resid, void* __restrict__ outp,
    int M, int N, int K)
{
    // single LDS symbol => known byte offsets for the TDM descriptors
    constexpr int NTILES = (EPI == EPI_GATED) ? 6 : 4;
    __shared__ alignas(16) bf16 smem[NTILES * TILE_ELEMS];
    bf16* const sA = smem;                    // bufs 0,1
    bf16* const sB = smem + 2 * TILE_ELEMS;   // bufs 0,1
    bf16* const sG = smem + 4 * TILE_ELEMS;   // bufs 0,1 (gated only)

    const int tid  = threadIdx.x;
    const int lane = tid & 31;
    const int wave = tid >> 5;
    const int wm   = wave & 3;          // 4 waves along M -> 32 rows each
    const int wn   = wave >> 2;         // 2 waves along N -> 64 cols each
    const int bn0  = blockIdx.x * BN;
    const int bm0  = blockIdx.y * BM;

    const bf16* Ab  = A  + (size_t)bm0 * K;
    const bf16* B1b = B1 + (size_t)bn0 * K;
    const bf16* B2b = (EPI == EPI_GATED) ? (B2 + (size_t)bn0 * K) : nullptr;

    v8f acc[2][4];
    constexpr int GA = (EPI == EPI_GATED) ? 2 : 1;
    constexpr int GB = (EPI == EPI_GATED) ? 4 : 1;
    v8f acc2[GA][GB];
    const v8f vzero = {0.f,0.f,0.f,0.f,0.f,0.f,0.f,0.f};
    #pragma unroll
    for (int i = 0; i < 2; ++i)
        #pragma unroll
        for (int j = 0; j < 4; ++j) acc[i][j] = vzero;
    #pragma unroll
    for (int i = 0; i < GA; ++i)
        #pragma unroll
        for (int j = 0; j < GB; ++j) acc2[i][j] = vzero;

    // Scalar (SGPR) branch: TDM ignores EXEC, so the guard must be a true
    // scalar branch or every wave would redundantly issue the DMA.
    const int wv = __builtin_amdgcn_readfirstlane(wave);
    if (wv == 0) {
        tdm_load_tile(Ab,  0u * TILE_BYTES, K);
        tdm_load_tile(B1b, 2u * TILE_BYTES, K);
        if constexpr (EPI == EPI_GATED) tdm_load_tile(B2b, 4u * TILE_BYTES, K);
    }
    __builtin_amdgcn_s_wait_tensorcnt(0);
    __syncthreads();

    const int nk = K / BK;
    for (int kt = 0; kt < nk; ++kt) {
        const int cur = kt & 1, nxt = cur ^ 1;
        if (kt + 1 < nk && wv == 0) {
            const int ko = (kt + 1) * BK;
            tdm_load_tile(Ab  + ko, (0u + (u32)nxt) * TILE_BYTES, K);
            tdm_load_tile(B1b + ko, (2u + (u32)nxt) * TILE_BYTES, K);
            if constexpr (EPI == EPI_GATED)
                tdm_load_tile(B2b + ko, (4u + (u32)nxt) * TILE_BYTES, K);
        }

        v16bf af[2];
        #pragma unroll
        for (int mi = 0; mi < 2; ++mi) af[mi] = fragA(sA + cur * TILE_ELEMS, wm * 32 + mi * 16, lane);
        v16bf bf_[4];
        #pragma unroll
        for (int ni = 0; ni < 4; ++ni) bf_[ni] = fragB(sB + cur * TILE_ELEMS, wn * 64 + ni * 16, lane);

        #pragma unroll
        for (int mi = 0; mi < 2; ++mi)
            #pragma unroll
            for (int ni = 0; ni < 4; ++ni)
                acc[mi][ni] = __builtin_amdgcn_wmma_f32_16x16x32_bf16(
                    false, af[mi], false, bf_[ni], (short)0, acc[mi][ni], false, false);

        if constexpr (EPI == EPI_GATED) {
            v16bf gf[4];
            #pragma unroll
            for (int ni = 0; ni < 4; ++ni)
                gf[ni] = fragB(sG + cur * TILE_ELEMS, wn * 64 + ni * 16, lane);
            #pragma unroll
            for (int mi = 0; mi < 2; ++mi)
                #pragma unroll
                for (int ni = 0; ni < 4; ++ni)
                    acc2[mi][ni] = __builtin_amdgcn_wmma_f32_16x16x32_bf16(
                        false, af[mi], false, gf[ni], (short)0, acc2[mi][ni], false, false);
        }

        __builtin_amdgcn_s_wait_tensorcnt(0);   // prefetch DMA done (no-op for waves 1..7)
        __syncthreads();
    }

    // Epilogue. C/D layout: element j -> row m0+j+8*(lane>=16), col n0+(lane&15)
    const int rowoff = (lane >> 4) * 8;
    #pragma unroll
    for (int mi = 0; mi < 2; ++mi) {
        #pragma unroll
        for (int ni = 0; ni < 4; ++ni) {
            const int nn    = bn0 + wn * 64 + ni * 16 + (lane & 15);
            const int mbase = bm0 + wm * 32 + mi * 16 + rowoff;
            const float b1 = bias1[nn];
            float b2 = 0.f;
            if constexpr (EPI == EPI_GATED) b2 = bias2[nn];
            #pragma unroll
            for (int j = 0; j < 8; ++j) {
                const size_t idx = (size_t)(mbase + j) * N + nn;
                const float v = acc[mi][ni][j] + b1;
                if constexpr (EPI == EPI_GATED) {
                    float g = acc2[mi][ni][j] + b2;
                    g = 1.0f / (1.0f + __expf(-g));
                    ((bf16*)outp)[idx] = f2bf(v * g);
                } else if constexpr (EPI == EPI_BF16) {
                    ((bf16*)outp)[idx] = f2bf(v);
                } else {
                    ((float*)outp)[idx] = v + resid[idx];
                }
            }
        }
    }
}

// ---------------- chunked linear scan: h_t = a*h_{t-1} + b_t --------------
__global__ __launch_bounds__(256) void k_scan_chunk(const bf16* __restrict__ bA,
                                                    const float* __restrict__ a_log,
                                                    float* __restrict__ chunk_h) {
    const int c  = 2 * (blockIdx.x * blockDim.x + threadIdx.x);  // 2 channels/thread
    const int ck = blockIdx.y, bt = blockIdx.z;
    const float a0 = 1.0f / (1.0f + __expf(a_log[c]));
    const float a1 = 1.0f / (1.0f + __expf(a_log[c + 1]));
    float h0 = 0.f, h1 = 0.f;
    const u32* bp = (const u32*)(bA + ((size_t)(bt * SEQ + ck * CHUNK)) * INNER_ + c);
    #pragma unroll 4
    for (int t = 0; t < CHUNK; ++t) {
        const u32 w = bp[(size_t)t * (INNER_ / 2)];
        h0 = a0 * h0 + bfbits2f(w & 0xFFFFu);
        h1 = a1 * h1 + bfbits2f(w >> 16);
    }
    float* o = chunk_h + ((size_t)(bt * NCHUNK + ck)) * INNER_ + c;
    o[0] = h0; o[1] = h1;
}

__global__ void k_scan_carry(const float* __restrict__ chunk_h,
                             const float* __restrict__ a_log,
                             float* __restrict__ carry) {
    const int idx = blockIdx.x * blockDim.x + threadIdx.x;
    if (idx >= NBATCH * INNER_) return;
    const int bt = idx / INNER_, c = idx % INNER_;
    const float a  = 1.0f / (1.0f + __expf(a_log[c]));
    const float aL = __powf(a, (float)CHUNK);
    float cur = 0.f;
    for (int k = 0; k < NCHUNK; ++k) {
        const size_t o = ((size_t)(bt * NCHUNK + k)) * INNER_ + c;
        carry[o] = cur;
        cur = aL * cur + chunk_h[o];
    }
}

__global__ __launch_bounds__(256) void k_scan_apply(const bf16* __restrict__ bA,
                                                    const bf16* __restrict__ cA,
                                                    const bf16* __restrict__ dA,
                                                    const float* __restrict__ a_log,
                                                    const float* __restrict__ carry,
                                                    bf16* __restrict__ y) {
    const int c  = 2 * (blockIdx.x * blockDim.x + threadIdx.x);
    const int ck = blockIdx.y, bt = blockIdx.z;
    const float a0 = 1.0f / (1.0f + __expf(a_log[c]));
    const float a1 = 1.0f / (1.0f + __expf(a_log[c + 1]));
    const size_t co = ((size_t)(bt * NCHUNK + ck)) * INNER_ + c;
    float h0 = carry[co], h1 = carry[co + 1];
    const size_t base = ((size_t)(bt * SEQ + ck * CHUNK)) * INNER_ + c;
    const u32* bp = (const u32*)(bA + base);
    const u32* cp = (const u32*)(cA + base);
    const u32* dp = (const u32*)(dA + base);
    u32* yp = (u32*)(y + base);
    #pragma unroll 4
    for (int t = 0; t < CHUNK; ++t) {
        const size_t off = (size_t)t * (INNER_ / 2);
        const u32 bw = bp[off], cw = cp[off], dw = dp[off];
        h0 = a0 * h0 + bfbits2f(bw & 0xFFFFu);
        h1 = a1 * h1 + bfbits2f(bw >> 16);
        const float y0 = bfbits2f(cw & 0xFFFFu) * h0 + bfbits2f(dw & 0xFFFFu);
        const float y1 = bfbits2f(cw >> 16) * h1 + bfbits2f(dw >> 16);
        yp[off] = (u32)f2bf_bits(y0) | ((u32)f2bf_bits(y1) << 16);
    }
}

// ---------------- driver ---------------------------------------------------
extern "C" void kernel_launch(void* const* d_in, const int* in_sizes, int n_in,
                              void* d_out, int out_size, void* d_ws, size_t ws_size,
                              hipStream_t stream) {
    const float* x      = (const float*)d_in[0];
    const float* norm_w = (const float*)d_in[1];
    const float* in_w   = (const float*)d_in[2];
    const float* in_b   = (const float*)d_in[3];
    const float* gate_w = (const float*)d_in[4];
    const float* gate_b = (const float*)d_in[5];
    const float* b_w    = (const float*)d_in[6];
    const float* b_b    = (const float*)d_in[7];
    const float* c_w    = (const float*)d_in[8];
    const float* c_b    = (const float*)d_in[9];
    const float* d_w    = (const float*)d_in[10];
    const float* d_b    = (const float*)d_in[11];
    const float* out_w  = (const float*)d_in[12];
    const float* out_b  = (const float*)d_in[13];
    const float* a_log  = (const float*)d_in[14];
    float* out = (float*)d_out;

    size_t off = 0;
    auto alloc = [&](size_t bytes) {
        void* p = (char*)d_ws + off;
        off += (bytes + 255) & ~(size_t)255;
        return p;
    };
    bf16* wIn   = (bf16*)alloc((size_t)INNER_ * DIM_   * 2);
    bf16* wGate = (bf16*)alloc((size_t)INNER_ * DIM_   * 2);
    bf16* wB    = (bf16*)alloc((size_t)INNER_ * INNER_ * 2);
    bf16* wC    = (bf16*)alloc((size_t)INNER_ * INNER_ * 2);
    bf16* wD    = (bf16*)alloc((size_t)INNER_ * INNER_ * 2);
    bf16* wOut  = (bf16*)alloc((size_t)DIM_   * INNER_ * 2);
    bf16* xn    = (bf16*)alloc((size_t)NTOK * DIM_   * 2);
    bf16* u     = (bf16*)alloc((size_t)NTOK * INNER_ * 2);
    bf16* bArr  = (bf16*)alloc((size_t)NTOK * INNER_ * 2);
    bf16* cArr  = (bf16*)alloc((size_t)NTOK * INNER_ * 2);
    bf16* dArr  = (bf16*)alloc((size_t)NTOK * INNER_ * 2);
    bf16* yArr  = (bf16*)alloc((size_t)NTOK * INNER_ * 2);
    float* chunk_h = (float*)alloc((size_t)NBATCH * NCHUNK * INNER_ * 4);
    float* carry   = (float*)alloc((size_t)NBATCH * NCHUNK * INNER_ * 4);

    auto cvt = [&](const float* src, bf16* dst, size_t n) {
        int n4 = (int)(n / 4);
        k_f32_to_bf16<<<(n4 + 255) / 256, 256, 0, stream>>>(src, dst, n4);
    };
    cvt(in_w,   wIn,   (size_t)INNER_ * DIM_);
    cvt(gate_w, wGate, (size_t)INNER_ * DIM_);
    cvt(b_w,    wB,    (size_t)INNER_ * INNER_);
    cvt(c_w,    wC,    (size_t)INNER_ * INNER_);
    cvt(d_w,    wD,    (size_t)INNER_ * INNER_);
    cvt(out_w,  wOut,  (size_t)DIM_ * INNER_);

    k_rmsnorm<<<NTOK, 256, 0, stream>>>(x, norm_w, xn);

    // u = (xn @ in_w^T + in_b) * sigmoid(xn @ gate_w^T + gate_b)
    k_gemm<EPI_GATED><<<dim3(INNER_ / BN, NTOK / BM), 256, 0, stream>>>(
        xn, wIn, wGate, in_b, gate_b, nullptr, u, NTOK, INNER_, DIM_);

    // b, c, d projections
    k_gemm<EPI_BF16><<<dim3(INNER_ / BN, NTOK / BM), 256, 0, stream>>>(
        u, wB, nullptr, b_b, nullptr, nullptr, bArr, NTOK, INNER_, INNER_);
    k_gemm<EPI_BF16><<<dim3(INNER_ / BN, NTOK / BM), 256, 0, stream>>>(
        u, wC, nullptr, c_b, nullptr, nullptr, cArr, NTOK, INNER_, INNER_);
    k_gemm<EPI_BF16><<<dim3(INNER_ / BN, NTOK / BM), 256, 0, stream>>>(
        u, wD, nullptr, d_b, nullptr, nullptr, dArr, NTOK, INNER_, INNER_);

    // chunked linear recurrence + y = c*h + d
    k_scan_chunk<<<dim3(INNER_ / 512, NCHUNK, NBATCH), 256, 0, stream>>>(bArr, a_log, chunk_h);
    k_scan_carry<<<(NBATCH * INNER_ + 255) / 256, 256, 0, stream>>>(chunk_h, a_log, carry);
    k_scan_apply<<<dim3(INNER_ / 512, NCHUNK, NBATCH), 256, 0, stream>>>(
        bArr, cArr, dArr, a_log, carry, yArr);

    // out = y @ out_w^T + out_b + residual
    k_gemm<EPI_RESID><<<dim3(DIM_ / BN, NTOK / BM), 256, 0, stream>>>(
        yArr, wOut, nullptr, out_b, nullptr, x, out, NTOK, DIM_, INNER_);
}
